// glimpse_module_28802050687321
// MI455X (gfx1250) — matrix operand order repaired and checked by reading, not compiled, not run on hardware
//
#include <hip/hip_runtime.h>

typedef __attribute__((ext_vector_type(16))) __bf16        v16bf;
typedef __attribute__((ext_vector_type(8)))  float         v8f;
typedef __attribute__((ext_vector_type(4)))  unsigned int  u32x4;
typedef __attribute__((ext_vector_type(4)))  float         f32x4;

union ABits {
  u32x4 u[2];
  v16bf v;
  unsigned short s[16];
};

__device__ __forceinline__ unsigned short f32_to_bf16(float f) {
  unsigned int u = __builtin_bit_cast(unsigned int, f);
  u += 0x7FFFu + ((u >> 16) & 1u);   // round-to-nearest-even
  return (unsigned short)(u >> 16);
}

// ---------------------------------------------------------------------------
// Prep: pack w2 (128,64,3,3) f32 into WMMA B-fragment order, bf16:
//   bfrag[nt][q][lane][v]  (uint32 = bf16 pair), nt<8, q<18, lane<32, v<8
//   kk = tap*64 + c1 ; chunk q covers kk in [32q,32q+32)
//   B layout: lane L -> col n = L%16 ; kk_local = (L/16)*16 + 2v + {0,1}
// ---------------------------------------------------------------------------
__global__ __launch_bounds__(256) void prep_bfrag(const float* __restrict__ w2,
                                                  unsigned int* __restrict__ bfrag) {
  int idx = blockIdx.x * 256 + threadIdx.x;        // [8][18][32][8]
  if (idx >= 8 * 18 * 32 * 8) return;
  int v    = idx & 7;
  int lane = (idx >> 3) & 31;
  int qq   = (idx >> 8) % 18;
  int nt   = (idx >> 8) / 18;
  int tap  = qq >> 1;
  int c1b  = (qq & 1) << 5;
  int dy = tap / 3, dx = tap % 3;
  int n  = (nt << 4) + (lane & 15);
  int kk0 = ((lane >> 4) << 4) + 2 * v;
  int ca = c1b + kk0, cb = c1b + kk0 + 1;
  float va = w2[((n * 64 + ca) * 3 + dy) * 3 + dx];
  float vb = w2[((n * 64 + cb) * 3 + dy) * 3 + dx];
  bfrag[idx] = (unsigned)f32_to_bf16(va) | ((unsigned)f32_to_bf16(vb) << 16);
}

// ---------------------------------------------------------------------------
// Fused: foveation gather -> conv1(3->64, WMMA) -> conv2(64->128, s2, WMMA)
// One block = (b, scale k, 16x16 conv2-output tile). 256 threads = 8 waves.
//
// h1 grid: 33x33 positions (exactly what conv2's 16x16/s2 tile consumes),
// position space padded to 69 M-tiles (1104 rows) so tail stores are
// unconditional.  ins has a zero-filled pseudo-channel c=3 so the K-padding
// (27->32) of conv1's im2col needs no lane-divergent guard.
// ---------------------------------------------------------------------------
__global__ __launch_bounds__(256) void glimpse_fused(
    const float* __restrict__ x, const float* __restrict__ l,
    const float* __restrict__ w1, const float* __restrict__ b1,
    const float* __restrict__ b2, const unsigned int* __restrict__ bfrag,
    float* __restrict__ out) {
  __shared__ __align__(16) unsigned short h1s[1104 * 64];   // 141312 B (bf16)
  __shared__ __align__(16) unsigned short ins[5148];        //  10296 B (3*36*36 + zero pad)
  __shared__ __align__(16) unsigned int   w1f[4 * 32 * 8];  //   4096 B
  __shared__ float b1s[64];                                 //    256 B

  const int tid  = threadIdx.x;
  const int lane = tid & 31;
  const int wv   = tid >> 5;       // wave id 0..7
  const int lh   = lane >> 4;      // lane half
  const int ln   = lane & 15;

  const int blk  = blockIdx.x;
  const int tile = blk & 3;
  const int bk   = blk >> 2;
  const int k    = bk % 3;
  const int b    = bk / 3;
  const int ty   = (tile >> 1) << 4;   // conv2 output tile origin (32x32 space)
  const int tx   = (tile & 1) << 4;

  // fixation -> pixel coords (clip(floor(0.5*(l+1)*512), 0, 512))
  const float ly = l[2 * b], lx = l[2 * b + 1];
  int cy = (int)floorf(0.5f * (ly + 1.0f) * 512.0f);
  int cx = (int)floorf(0.5f * (lx + 1.0f) * 512.0f);
  cy = cy < 0 ? 0 : (cy > 512 ? 512 : cy);
  cx = cx < 0 ? 0 : (cx > 512 ? 512 : cx);

  const int step = 1 << k;       // nearest-resize stride in source pixels
  const int half = 32 << k;      // patch_size/2
  const int hy0 = 2 * ty - 1;    // h1 tile origin in 64x64 patch space
  const int hx0 = 2 * tx - 1;

  if (tid < 64) b1s[tid] = b1[tid];

  // w1 B-fragments: lane L -> col c1 = L%16 ; kk = (L/16)*16 + 2v + {0,1}; pad K 27->32
  for (int i = tid; i < 4 * 32 * 8; i += 256) {
    int v = i & 7, li = (i >> 3) & 31, ntl = i >> 8;
    int c1  = (ntl << 4) + (li & 15);
    int kk0 = ((li >> 4) << 4) + 2 * v;
    float a = (kk0     < 27) ? w1[c1 * 27 + kk0]     : 0.0f;
    float c = (kk0 + 1 < 27) ? w1[c1 * 27 + kk0 + 1] : 0.0f;
    w1f[i] = (unsigned)f32_to_bf16(a) | ((unsigned)f32_to_bf16(c) << 16);
  }

  // foveated gather of the 36x36x3 halo (zero pad outside patch / image),
  // plus zero-filled pseudo-channel region [3888, 5148) for K-padding.
  const float* xb = x + (size_t)b * (3 * 512 * 512);
  for (int i = tid; i < 5148; i += 256) {
    float v = 0.0f;
    if (i < 3888) {
      int c = i / 1296, rr = i % 1296;
      int iy = rr / 36, ix = rr % 36;
      int py = hy0 - 1 + iy, px = hx0 - 1 + ix;   // patch coords
      if (((unsigned)py < 64u) && ((unsigned)px < 64u)) {
        int sy = cy - half + py * step;
        int sx = cx - half + px * step;
        if (((unsigned)sy < 512u) && ((unsigned)sx < 512u))
          v = xb[(size_t)c * 262144 + sy * 512 + sx];
      }
    }
    ins[i] = f32_to_bf16(v);
  }
  __syncthreads();

  // ---- conv1 via WMMA: M = 1089 positions (69 tiles of 16), N = 64, K = 27->32
  for (int t = wv; t < 276; t += 8) {
    int ptile = t >> 2;
    int ntl   = t & 3;
    int c1    = (ntl << 4) + ln;

    ABits bfr;
    const u32x4* bp = (const u32x4*)&w1f[(((ntl << 5) + lane) << 3)];
    bfr.u[0] = bp[0]; bfr.u[1] = bp[1];

    int m = ptile * 16 + ln;
    int p = m < 1089 ? m : 1088;     // clamp tail-tile reads
    int r = p / 33, cc = p % 33;

    ABits afr;                        // im2col gather, A lane layout (branchless)
#pragma unroll
    for (int h = 0; h < 16; ++h) {
      int v = h >> 1, hl2 = h & 1;
      int kk = (lh << 4) + h;
      int c = kk / 9, rm = kk % 9;    // kk>=27 -> c==3 -> zero region
      afr.s[h] = ins[c * 1296 + (r + rm / 3) * 36 + (cc + rm % 3)];
    }

    v8f acc;
    float bias = b1s[c1];
#pragma unroll
    for (int rr = 0; rr < 8; ++rr) acc[rr] = bias;
    acc = __builtin_amdgcn_wmma_f32_16x16x32_bf16(false, afr.v, false, bfr.v,
                                                  (short)0, acc, false, false);
#pragma unroll
    for (int rr = 0; rr < 8; ++rr) {
      int pp = ptile * 16 + rr + (lh << 3);      // <= 1103, always in bounds
      h1s[pp * 64 + c1] = f32_to_bf16(fmaxf(acc[rr], 0.0f));
    }
  }
  __syncthreads();

  // ---- conv2 via WMMA: M = 256 spatial (16 mt), N = 128 (wave = 16-ch tile), K = 576
  const int nt = wv;
  const int n  = (nt << 4) + ln;
  const float bias2 = b2[n];
  const unsigned int* bq = bfrag + ((((nt * 18) << 5) + lane) << 3);

  for (int mt = 0; mt < 16; ++mt) {
    v8f acc;
#pragma unroll
    for (int rr = 0; rr < 8; ++rr) acc[rr] = bias2;
#pragma unroll
    for (int q = 0; q < 18; ++q) {
      const int tap = q >> 1;
      const int dy = tap / 3, dx = tap % 3;
      const int chb  = ((q & 1) << 5) + (lh << 4);
      const int base = ((2 * mt + dy) * 33 + (2 * ln + dx)) * 64 + chb;
      ABits afr;                                   // 2x ds_load_b128
      afr.u[0] = *(const u32x4*)&h1s[base];
      afr.u[1] = *(const u32x4*)&h1s[base + 8];
      ABits bfr;                                   // 2x global_load_b128 (L2-resident)
      const u32x4* bp = (const u32x4*)(bq + q * 256);
      bfr.u[0] = bp[0]; bfr.u[1] = bp[1];
      acc = __builtin_amdgcn_wmma_f32_16x16x32_bf16(false, afr.v, false, bfr.v,
                                                    (short)0, acc, false, false);
    }
    const int oy = ty + mt;
    float* ob = out + ((((size_t)b * 3 + k) * 128 + n) * 32 + oy) * 32 + tx + (lh << 3);
    f32x4 s0, s1;                                  // 2x global_store_b128 per lane
#pragma unroll
    for (int rr = 0; rr < 4; ++rr) {
      s0[rr] = fmaxf(acc[rr], 0.0f);
      s1[rr] = fmaxf(acc[rr + 4], 0.0f);
    }
    *(f32x4*)ob       = s0;
    *(f32x4*)(ob + 4) = s1;
  }
}

// ---------------------------------------------------------------------------
extern "C" void kernel_launch(void* const* d_in, const int* in_sizes, int n_in,
                              void* d_out, int out_size, void* d_ws, size_t ws_size,
                              hipStream_t stream) {
  const float* x  = (const float*)d_in[0];
  const float* l  = (const float*)d_in[1];
  const float* w1 = (const float*)d_in[2];
  const float* b1 = (const float*)d_in[3];
  const float* w2 = (const float*)d_in[4];
  const float* b2 = (const float*)d_in[5];
  float* out = (float*)d_out;
  unsigned int* bfrag = (unsigned int*)d_ws;   // 8*18*32*8 u32 = 147456 B

  prep_bfrag<<<(8 * 18 * 32 * 8 + 255) / 256, 256, 0, stream>>>(w2, bfrag);
  glimpse_fused<<<64 * 3 * 4, 256, 0, stream>>>(x, l, w1, b1, b2, bfrag, out);
}
